// GraphNet_86363202388241
// MI455X (gfx1250) — matrix operand order repaired and checked by reading, not compiled, not run on hardware
//
#include <hip/hip_runtime.h>
#include <hip/hip_bf16.h>
#include <cstddef>

// ---------------------------------------------------------------------------
// Types for CDNA5 WMMA (wave32): v_wmma_f32_16x16x32_bf16
// ---------------------------------------------------------------------------
typedef __attribute__((ext_vector_type(16))) __bf16        v16bf;
typedef __attribute__((ext_vector_type(8)))  float          v8f;
typedef __attribute__((ext_vector_type(8)))  unsigned short u16x8;

union ABFrag {
    struct { u16x8 lo, hi; } p;   // two 16B halves (VGPRs 0-3 / 4-7)
    v16bf v;
};

__device__ __forceinline__ unsigned short f32_to_bf16(float f) {
    unsigned u = __float_as_uint(f);
    u += 0x7FFFu + ((u >> 16) & 1u);          // round-to-nearest-even
    return (unsigned short)(u >> 16);
}

// Monotone order-preserving fp32 <-> u32 mapping (for atomicMax on floats)
__device__ __forceinline__ unsigned enc_f32(float f) {
    unsigned u = __float_as_uint(f);
    return (u & 0x80000000u) ? ~u : (u | 0x80000000u);
}
__device__ __forceinline__ float dec_f32(unsigned k) {
    unsigned u = (k & 0x80000000u) ? (k & 0x7FFFFFFFu) : ~k;
    return __uint_as_float(u);
}

// ---------------------------------------------------------------------------
// f32 -> bf16 conversion / fills
// ---------------------------------------------------------------------------
__global__ void k_conv_bf16(const float* __restrict__ s,
                            unsigned short* __restrict__ d, size_t n) {
    size_t i = (size_t)blockIdx.x * blockDim.x + threadIdx.x;
    if (i < n) d[i] = f32_to_bf16(s[i]);
}

__global__ void k_fill_u32(unsigned* __restrict__ p, unsigned v, size_t n) {
    size_t i = (size_t)blockIdx.x * blockDim.x + threadIdx.x;
    if (i < n) p[i] = v;
}

// ---------------------------------------------------------------------------
// WMMA bf16 GEMM:  C[M,Nc] = A[M,K] * B[K,Nc]  (+bias, +relu via template)
// Block: 256 thr = 8 waves, block tile 64(M) x 64(N), K-step 32.
// Wave w owns TWO stacked 16x16 tiles at rows (w/4)*32 and (w/4)*32+16,
// cols (w%4)*16 — the B fragment is reused by both WMMAs (back-to-back issue).
// Staging: one coalesced b128 global load per thread for A and for B;
// LDS rows padded to 80B (5x16B) -> 16B-aligned, bank-conflict-free b128 reads.
// ---------------------------------------------------------------------------
#define GBM 64
#define GBN 64
#define GBK 32
#define LPAD 40

template<bool BIAS, bool RELU>
__global__ __launch_bounds__(256) void k_gemm_wmma(
    const unsigned short* __restrict__ A,   // bf16 [M,K]
    const unsigned short* __restrict__ B,   // bf16 [K,Nc]
    float* __restrict__ C,                  // f32 [M,Nc]
    int M, int K, int Nc,
    const float* __restrict__ bias)
{
    __shared__ __align__(16) unsigned short Alds[GBM][LPAD];
    __shared__ __align__(16) unsigned short Blds[GBN][LPAD];  // transposed [n][k]

    const int tid  = threadIdx.x;
    const int wave = tid >> 5;
    const int lane = tid & 31;
    const int m0 = blockIdx.x * GBM;
    const int n0 = blockIdx.y * GBN;
    const int mt = (wave >> 2) * 32;      // wave covers rows mt.. and mt+16..
    const int nt = (wave & 3) * 16;
    const int hl  = lane >> 4;            // lane half
    const int l16 = lane & 15;

    // Staging coordinates: one 16B load per thread per block
    const int ar = tid >> 2, ac = (tid & 3) * 8;   // A: 64 rows x 4 chunks of 8
    const int bk = tid >> 3, bn = (tid & 7) * 8;   // B: 32 k-rows x 8 chunks of 8

    v8f acc0 = {}, acc1 = {};

#pragma unroll 2
    for (int k0 = 0; k0 < K; k0 += GBK) {
        // Stage A (64x32): guarded coalesced b128 load, zeros past M
        {
            int gm = m0 + ar;
            u16x8 av = {};
            if (gm < M) av = *(const u16x8*)(A + (size_t)gm * K + k0 + ac);
            *(u16x8*)&Alds[ar][ac] = av;
        }
        // Stage B (32x64) transposed into [n][k]
        {
            u16x8 bv = *(const u16x8*)(B + (size_t)(k0 + bk) * Nc + n0 + bn);
#pragma unroll
            for (int j = 0; j < 8; ++j) Blds[bn + j][bk] = bv[j];
        }
        __syncthreads();

        // Fragments per ISA 16-bit layouts (A 16x32: K-octet by lane half;
        // B 32x16: column per lane, same K-octet order as A)
        ABFrag a0, a1, b;
        a0.p.lo = *(const u16x8*)&Alds[mt + l16][hl * 8];
        a0.p.hi = *(const u16x8*)&Alds[mt + l16][16 + hl * 8];
        a1.p.lo = *(const u16x8*)&Alds[mt + 16 + l16][hl * 8];
        a1.p.hi = *(const u16x8*)&Alds[mt + 16 + l16][16 + hl * 8];
        b.p.lo  = *(const u16x8*)&Blds[nt + l16][hl * 8];
        b.p.hi  = *(const u16x8*)&Blds[nt + l16][16 + hl * 8];

        acc0 = __builtin_amdgcn_wmma_f32_16x16x32_bf16(
            false, a0.v, false, b.v, (short)0, acc0, false, false);
        acc1 = __builtin_amdgcn_wmma_f32_16x16x32_bf16(
            false, a1.v, false, b.v, (short)0, acc1, false, false);
        __syncthreads();
    }

    // Epilogue: bias depends only on the lane's column -> load once
    const int n = n0 + nt + l16;
    const float bv = BIAS ? bias[n] : 0.0f;
#pragma unroll
    for (int r = 0; r < 8; ++r) {
        int m = m0 + mt + r + 8 * hl;
        if (m < M) {
            float v = acc0[r] + bv;
            if (RELU) v = fmaxf(v, 0.0f);
            C[(size_t)m * Nc + n] = v;
        }
        int m2 = m + 16;
        if (m2 < M) {
            float v = acc1[r] + bv;
            if (RELU) v = fmaxf(v, 0.0f);
            C[(size_t)m2 * Nc + n] = v;
        }
    }
}

// ---------------------------------------------------------------------------
// Attention coefficients: a_s[n,h] = <h[n,h,:], att_src[h,:]>, same for a_d
// ---------------------------------------------------------------------------
__global__ void k_att(const float* __restrict__ h,
                      const float* __restrict__ att_src,
                      const float* __restrict__ att_dst,
                      float* __restrict__ a_s, float* __restrict__ a_d, int Nn) {
    int id = blockIdx.x * blockDim.x + threadIdx.x;
    if (id >= Nn * 4) return;
    int n = id >> 2, hd = id & 3;
    const float* hp = h + (size_t)n * 256 + hd * 64;
    const float* as = att_src + hd * 64;
    const float* ad = att_dst + hd * 64;
    float s = 0.f, d = 0.f;
#pragma unroll 8
    for (int c = 0; c < 64; ++c) { float v = hp[c]; s += v * as[c]; d += v * ad[c]; }
    a_s[id] = s; a_d[id] = d;
}

// ---------------------------------------------------------------------------
// Edge pass 1: segment max of leaky_relu(a_s[src]+a_d[dst]) via u32 atomicMax
// ---------------------------------------------------------------------------
__global__ void k_edge_max(const int* __restrict__ src, const int* __restrict__ dst,
                           const float* __restrict__ a_s, const float* __restrict__ a_d,
                           unsigned* __restrict__ mxu, int E, int Nn) {
    int id = blockIdx.x * blockDim.x + threadIdx.x;
    int total = (E + Nn) * 4;
    if (id >= total) return;
    int e = id >> 2, hd = id & 3;
    int s, d;
    if (e < E) { s = src[e]; d = dst[e]; } else { s = d = e - E; }
    float v = a_s[s * 4 + hd] + a_d[d * 4 + hd];
    v = (v > 0.f) ? v : 0.2f * v;                 // leaky_relu, slope 0.2
    atomicMax(&mxu[d * 4 + hd], enc_f32(v));
}

// ---------------------------------------------------------------------------
// Edge pass 2: ex = exp(e - max), store per-edge, atomicAdd denom
// ---------------------------------------------------------------------------
__global__ void k_edge_exp(const int* __restrict__ src, const int* __restrict__ dst,
                           const float* __restrict__ a_s, const float* __restrict__ a_d,
                           const unsigned* __restrict__ mxu,
                           float* __restrict__ exbuf, float* __restrict__ dn,
                           int E, int Nn) {
    int id = blockIdx.x * blockDim.x + threadIdx.x;
    int total = (E + Nn) * 4;
    if (id >= total) return;
    int e = id >> 2, hd = id & 3;
    int s, d;
    if (e < E) { s = src[e]; d = dst[e]; } else { s = d = e - E; }
    float v = a_s[s * 4 + hd] + a_d[d * 4 + hd];
    v = (v > 0.f) ? v : 0.2f * v;
    float ex = __expf(v - dec_f32(mxu[d * 4 + hd]));
    exbuf[id] = ex;
    atomicAdd(&dn[d * 4 + hd], ex);
}

// ---------------------------------------------------------------------------
// Edge pass 3: wave per edge; lane l handles channels [8l, 8l+8), head = l/8.
// float4 gathers of h[src], 8 fp32 atomics into agg[dst] (hits 192MB L2).
// ---------------------------------------------------------------------------
__global__ __launch_bounds__(256) void k_edge_aggr(
    const int* __restrict__ src, const int* __restrict__ dst,
    const float* __restrict__ h, const float* __restrict__ exbuf,
    const float* __restrict__ dn, float* __restrict__ agg, int E, int Nn) {
    int gwave  = (int)(((size_t)blockIdx.x * blockDim.x + threadIdx.x) >> 5);
    int lane   = threadIdx.x & 31;
    int total  = E + Nn;
    if (gwave >= total) return;
    int e = gwave;
    int s, d;
    if (e < E) { s = src[e]; d = dst[e]; } else { s = d = e - E; }
    int hd = lane >> 3;
    float alpha = exbuf[(size_t)e * 4 + hd] / (dn[d * 4 + hd] + 1e-16f);
    const float4* h4 = (const float4*)(h + (size_t)s * 256 + lane * 8);
    float* ag = agg + (size_t)d * 256 + lane * 8;
    float4 v0 = h4[0], v1 = h4[1];
    atomicAdd(ag + 0, v0.x * alpha); atomicAdd(ag + 1, v0.y * alpha);
    atomicAdd(ag + 2, v0.z * alpha); atomicAdd(ag + 3, v0.w * alpha);
    atomicAdd(ag + 4, v1.x * alpha); atomicAdd(ag + 5, v1.y * alpha);
    atomicAdd(ag + 6, v1.z * alpha); atomicAdd(ag + 7, v1.w * alpha);
}

// ---------------------------------------------------------------------------
// Finalize layer: bias (+relu), emit bf16 for the next GEMM input
// ---------------------------------------------------------------------------
__global__ void k_finalize(const float* __restrict__ agg, const float* __restrict__ bias,
                           int relu, unsigned short* __restrict__ outb, size_t total) {
    size_t id = (size_t)blockIdx.x * blockDim.x + threadIdx.x;
    if (id >= total) return;
    float v = agg[id] + bias[id & 255];
    if (relu) v = fmaxf(v, 0.0f);
    outb[id] = f32_to_bf16(v);
}

// ---------------------------------------------------------------------------
// Tiny FC2: out[n, 0..3] = fc1[n,:] @ Wc2 + bc2
// ---------------------------------------------------------------------------
__global__ void k_fc2(const float* __restrict__ fc1, const float* __restrict__ Wc2,
                      const float* __restrict__ bc2, float* __restrict__ out, int Nn) {
    int n = blockIdx.x * blockDim.x + threadIdx.x;
    if (n >= Nn) return;
    float o0 = bc2[0], o1 = bc2[1], o2 = bc2[2], o3 = bc2[3];
    const float* f = fc1 + (size_t)n * 64;
#pragma unroll 8
    for (int k = 0; k < 64; ++k) {
        float v = f[k];
        o0 += v * Wc2[k * 4 + 0]; o1 += v * Wc2[k * 4 + 1];
        o2 += v * Wc2[k * 4 + 2]; o3 += v * Wc2[k * 4 + 3];
    }
    float* o = out + (size_t)n * 4;
    o[0] = o0; o[1] = o1; o[2] = o2; o[3] = o3;
}

// ---------------------------------------------------------------------------
// Host orchestration
// ---------------------------------------------------------------------------
static inline int cdiv_ll(long long a, long long b) { return (int)((a + b - 1) / b); }

extern "C" void kernel_launch(void* const* d_in, const int* in_sizes, int n_in,
                              void* d_out, int out_size, void* d_ws, size_t ws_size,
                              hipStream_t stream) {
    const float* x   = (const float*)d_in[0];
    const int*   ei  = (const int*)d_in[1];
    const float* W1  = (const float*)d_in[2];
    const float* as1 = (const float*)d_in[3];
    const float* ad1 = (const float*)d_in[4];
    const float* b1  = (const float*)d_in[5];
    const float* W2  = (const float*)d_in[6];
    const float* as2 = (const float*)d_in[7];
    const float* ad2 = (const float*)d_in[8];
    const float* b2  = (const float*)d_in[9];
    const float* Wc1 = (const float*)d_in[10];
    const float* bc1 = (const float*)d_in[11];
    const float* Wc2 = (const float*)d_in[12];
    const float* bc2 = (const float*)d_in[13];
    float* out = (float*)d_out;

    const int Nn = in_sizes[0] / 256;
    const int E  = in_sizes[1] / 2;
    const int ET = E + Nn;                 // with self-loops
    const int* src = ei;
    const int* dst = ei + E;

    // Workspace carve-up (256B aligned slices)
    char* w = (char*)d_ws;
    auto take = [&](size_t bytes) -> void* {
        void* p = (void*)w;
        w += (bytes + 255) & ~(size_t)255;
        return p;
    };
    float*          h     = (float*)take((size_t)Nn * 256 * 4);   // pre-agg features
    float*          agg   = (float*)take((size_t)Nn * 256 * 4);   // aggregated out
    unsigned*       mxu   = (unsigned*)take((size_t)Nn * 4 * 4);  // segment max (encoded)
    float*          dn    = (float*)take((size_t)Nn * 4 * 4);     // softmax denom
    float*          a_s   = (float*)take((size_t)Nn * 4 * 4);
    float*          a_d   = (float*)take((size_t)Nn * 4 * 4);
    float*          exbuf = (float*)take((size_t)ET * 4 * 4);     // per-(edge,head) exp
    unsigned short* xb    = (unsigned short*)take((size_t)Nn * 256 * 2); // bf16 acts
    unsigned short* wb    = (unsigned short*)take((size_t)256 * 256 * 2);// bf16 weights
    float*          fc1   = exbuf;         // alias: exbuf dead by FC stage

    const size_t nodeElems = (size_t)Nn * 256;
    const dim3 gGemm(cdiv_ll(Nn, GBM), 256 / GBN);   // (782, 4) for Nc=256
    const dim3 gGemmFC(cdiv_ll(Nn, GBM), 1);         // Nc=64
    const int bEdgeH = cdiv_ll((long long)ET * 4, 256);
    const int bAggr  = cdiv_ll((long long)ET * 32, 256);
    const int bNode  = cdiv_ll(nodeElems, 256);
    const int bAtt   = cdiv_ll((long long)Nn * 4, 256);
    const int bNH    = cdiv_ll((long long)Nn * 4, 256);

    // ---- Layer 1 ----
    k_conv_bf16<<<bNode, 256, 0, stream>>>(x, xb, nodeElems);
    k_conv_bf16<<<cdiv_ll(65536, 256), 256, 0, stream>>>(W1, wb, 65536);
    k_gemm_wmma<false, false><<<gGemm, 256, 0, stream>>>(xb, wb, h, Nn, 256, 256, nullptr);
    k_att<<<bAtt, 256, 0, stream>>>(h, as1, ad1, a_s, a_d, Nn);
    k_fill_u32<<<bNH, 256, 0, stream>>>(mxu, 0u, (size_t)Nn * 4);
    k_fill_u32<<<bNH, 256, 0, stream>>>((unsigned*)dn, 0u, (size_t)Nn * 4);
    k_fill_u32<<<bNode, 256, 0, stream>>>((unsigned*)agg, 0u, nodeElems);
    k_edge_max<<<bEdgeH, 256, 0, stream>>>(src, dst, a_s, a_d, mxu, E, Nn);
    k_edge_exp<<<bEdgeH, 256, 0, stream>>>(src, dst, a_s, a_d, mxu, exbuf, dn, E, Nn);
    k_edge_aggr<<<bAggr, 256, 0, stream>>>(src, dst, h, exbuf, dn, agg, E, Nn);
    k_finalize<<<bNode, 256, 0, stream>>>(agg, b1, 1, xb, nodeElems);   // +relu

    // ---- Layer 2 ----
    k_conv_bf16<<<cdiv_ll(65536, 256), 256, 0, stream>>>(W2, wb, 65536);
    k_gemm_wmma<false, false><<<gGemm, 256, 0, stream>>>(xb, wb, h, Nn, 256, 256, nullptr);
    k_att<<<bAtt, 256, 0, stream>>>(h, as2, ad2, a_s, a_d, Nn);
    k_fill_u32<<<bNH, 256, 0, stream>>>(mxu, 0u, (size_t)Nn * 4);
    k_fill_u32<<<bNH, 256, 0, stream>>>((unsigned*)dn, 0u, (size_t)Nn * 4);
    k_fill_u32<<<bNode, 256, 0, stream>>>((unsigned*)agg, 0u, nodeElems);
    k_edge_max<<<bEdgeH, 256, 0, stream>>>(src, dst, a_s, a_d, mxu, E, Nn);
    k_edge_exp<<<bEdgeH, 256, 0, stream>>>(src, dst, a_s, a_d, mxu, exbuf, dn, E, Nn);
    k_edge_aggr<<<bAggr, 256, 0, stream>>>(src, dst, h, exbuf, dn, agg, E, Nn);
    k_finalize<<<bNode, 256, 0, stream>>>(agg, b2, 0, xb, nodeElems);   // no relu

    // ---- MLP head ----
    k_conv_bf16<<<cdiv_ll(256 * 64, 256), 256, 0, stream>>>(Wc1, wb, 256 * 64);
    k_gemm_wmma<true, true><<<gGemmFC, 256, 0, stream>>>(xb, wb, fc1, Nn, 256, 64, bc1);
    k_fc2<<<cdiv_ll(Nn, 256), 256, 0, stream>>>(fc1, Wc2, bc2, out, Nn);
}
